// Decoder_67920612819313
// MI455X (gfx1250) — matrix operand order repaired and checked by reading, not compiled, not run on hardware
//
#include <hip/hip_runtime.h>
#include <hip/hip_bf16.h>
#include <math.h>

// Problem dims (match reference)
#define B        8
#define T        384
#define IDIM     512
#define ODIM     80
#define L        192
#define DUNITS   1024
#define PRENET_D 256
#define ADIM     128
#define ACONV_CH 32
#define ACONV_K  31
#define POST_CH  512
#define POST_K   5
#define ZONEOUT  0.1f

typedef __attribute__((ext_vector_type(16))) __bf16 bf16x16;
typedef __attribute__((ext_vector_type(8)))  float  floatx8;

// ---------------- fragment layout helpers (CDNA5 WMMA bf16 16x16x32) --------
// A (16x32, MxK): lane l holds row m=l&15; lanes 0-15 carry K {0..7,16..23},
// lanes 16-31 carry K {8..15,24..31}; 16 bf16 per lane.
__device__ __forceinline__ int a_frag_k(int lane, int j) {
  int base = (lane & 16) ? 8 : 0;
  return (j < 8) ? (base + j) : (16 + base + (j - 8));
}
// B (32x16, KxN): lane l holds col n=l&15; lanes 0-15 carry K {0..7,16..23},
// lanes 16-31 carry K {8..15,24..31} (element j -> K).
__device__ __forceinline__ int b_frag_k(int lane, int j) {
  int base = (lane & 16) ? 8 : 0;
  return (j < 8) ? (base + j) : (16 + base + (j - 8));
}

__device__ __forceinline__ unsigned short f2bf(float f) {
  union { float f; unsigned u; } x; x.f = f;
  unsigned u = x.u;
  u += 0x7FFFu + ((u >> 16) & 1u);   // round-to-nearest-even
  return (unsigned short)(u >> 16);
}

// ---------------- weight packing: W (N,K) f32 -> fragment-contiguous bf16 ---
// dst[((nt*Ktiles + kt)*32 + lane)*16 + j]; two K-concatenated sources.
__global__ void pack_w(const float* w0, int K0, const float* w1, int K1,
                       int Nvalid, int Ktiles, unsigned short* dst) {
  int kt = blockIdx.x, nt = blockIdx.y, lane = threadIdx.x;
  int n = nt * 16 + (lane & 15);
  size_t o = ((size_t)(nt * Ktiles + kt) * 32 + lane) * 16;
  for (int j = 0; j < 16; ++j) {
    int k = kt * 32 + b_frag_k(lane, j);
    float v = 0.f;
    if (n < Nvalid) {
      if (k < K0)            v = w0[(size_t)n * K0 + k];
      else if (k < K0 + K1)  v = w1[(size_t)n * K1 + (k - K0)];
    }
    dst[o + j] = f2bf(v);
  }
}

// ---------------- activation packing: up to 3 concatenated (8,len) sources --
// Single M tile, rows 8..15 zero-padded.
__global__ void pack_act8(const float* s0, int len0, int ld0,
                          const float* s1, int len1, int ld1,
                          const float* s2, int len2, int ld2,
                          unsigned short* dst, int zeroAll) {
  int kt = blockIdx.x, lane = threadIdx.x;
  int m = lane & 15;
  size_t o = ((size_t)kt * 32 + lane) * 16;
  for (int j = 0; j < 16; ++j) {
    int k = kt * 32 + a_frag_k(lane, j);
    float v = 0.f;
    if (!zeroAll && m < B) {
      if (k < len0)                        v = s0[(size_t)m * ld0 + k];
      else if (k < len0 + len1)            v = s1[(size_t)m * ld1 + (k - len0)];
      else if (k < len0 + len1 + len2)     v = s2[(size_t)m * ld2 + (k - len0 - len1)];
    }
    dst[o + j] = f2bf(v);
  }
}

// ---------------- implicit-GEMM packing for postnet conv (SAME, K=5) --------
// act: (B*L, CI) f32.  K index kk = ci*5 + tap.  M rows = b*L + l.
__global__ void pack_conv_act(const float* act, int CI, int Ktiles,
                              unsigned short* dst) {
  int kt = blockIdx.x, mt = blockIdx.y, lane = threadIdx.x;
  int m = mt * 16 + (lane & 15);
  int b = m / L, l = m % L;
  size_t o = ((size_t)(mt * Ktiles + kt) * 32 + lane) * 16;
  int Kv = CI * POST_K;
  for (int j = 0; j < 16; ++j) {
    int k = kt * 32 + a_frag_k(lane, j);
    float v = 0.f;
    if (k < Kv) {
      int ci = k / POST_K, tap = k % POST_K;
      int ls = l + tap - POST_K / 2;
      if (ls >= 0 && ls < L) v = act[((size_t)b * L + ls) * CI + ci];
    }
    dst[o + j] = f2bf(v);
  }
}

// ---------------- epilogue shared by both GEMM variants ---------------------
__device__ __forceinline__ void gemm_epilogue(floatx8 c, int nt, int mt, int lane,
                                              const float* bias, const float* scale,
                                              const float* shift, float* out, int ldo,
                                              int Mvalid, int Nvalid, int act) {
  int n = nt * 16 + (lane & 15);
  if (n >= Nvalid) return;
  int rbase = (lane & 16) ? 8 : 0;  // D layout: lane<16 rows 0-7, else 8-15
  for (int r = 0; r < 8; ++r) {
    int m = mt * 16 + rbase + r;
    if (m >= Mvalid) continue;
    float v = c[r];
    if (bias)  v += bias[n];
    if (scale) v = scale[n] * v + shift[n];
    if (act == 1)      v = fmaxf(v, 0.f);
    else if (act == 2) v = tanhf(v);
    out[(size_t)m * ldo + n] = v;
  }
}

// ---------------- WMMA GEMM, weights streamed by per-lane global loads ------
__global__ void gemm_wmma(const unsigned short* Ap, const unsigned short* Wp,
                          const float* bias, const float* scale, const float* shift,
                          float* out, int ldo, int Ktiles, int Mvalid, int Nvalid,
                          int act) {
  int nt = blockIdx.x, mt = blockIdx.y, lane = threadIdx.x;
  floatx8 c = {};
  const unsigned short* ap = Ap + ((size_t)mt * Ktiles * 32 + lane) * 16;
  const unsigned short* bp = Wp + ((size_t)nt * Ktiles * 32 + lane) * 16;
  for (int kt = 0; kt < Ktiles; ++kt) {
    bf16x16 a = *(const bf16x16*)ap;
    bf16x16 b = *(const bf16x16*)bp;
    __builtin_prefetch(bp + 512, 0, 1);   // next weight tile -> global_prefetch_b8
    c = __builtin_amdgcn_wmma_f32_16x16x32_bf16(false, a, false, b,
                                                (short)0, c, false, false);
    ap += 512; bp += 512;
  }
  gemm_epilogue(c, nt, mt, lane, bias, scale, shift, out, ldo, Mvalid, Nvalid, act);
}

// ---------------- WMMA GEMM, weights streamed by the Tensor Data Mover ------
#if __has_builtin(__builtin_amdgcn_tensor_load_to_lds) && \
    __has_builtin(__builtin_amdgcn_s_wait_tensorcnt)
#define HAVE_TDM 1
typedef __attribute__((ext_vector_type(4))) unsigned int uint4v;
typedef __attribute__((ext_vector_type(8))) int int8v;
typedef __attribute__((ext_vector_type(4))) int int4v;

// 1-D TDM copy: global (bytes, 8B-aligned) -> LDS offset.  D# per ISA ch.8:
// group0 = {count=1 | lds_addr | global_addr[56:0] | type=2}
// group1 = {data_size=8B | tensor_dim0=tile_dim0=bytes/8 | dim1=1 | stride0=dim0}
__device__ __forceinline__ void tdm_copy_1d(const void* gsrc, unsigned ldsoff,
                                            unsigned bytes) {
  unsigned long long ga = (unsigned long long)gsrc;
  unsigned n8 = bytes >> 3;
  uint4v g0;
  g0[0] = 1u;                                            // count=1 (user D#)
  g0[1] = ldsoff;                                        // lds_addr
  g0[2] = (unsigned)(ga & 0xFFFFFFFFull);                // global_addr[31:0]
  g0[3] = (unsigned)((ga >> 32) & 0x01FFFFFFull) | 0x80000000u; // [56:32]|type=2
  int8v g1;
  g1[0] = 0x00030000;                                    // data_size=3 (8B)
  g1[1] = (int)((n8 & 0xFFFFu) << 16);                   // tensor_dim0 lo16
  g1[2] = (int)(((n8 >> 16) & 0xFFFFu) | (1u << 16));    // dim0 hi16 | dim1=1
  g1[3] = (int)((n8 & 0xFFFFu) << 16);                   // tile_dim0
  g1[4] = 0;                                             // tile_dim1/2 unused
  g1[5] = (int)n8;                                       // dim0_stride lo32
  g1[6] = 0;
  g1[7] = 0;
  int4v z = {0, 0, 0, 0};
#if __clang_major__ >= 23
  int8v z8 = {0, 0, 0, 0, 0, 0, 0, 0};
  __builtin_amdgcn_tensor_load_to_lds(g0, g1, z, z, z8, 0);
#else
  __builtin_amdgcn_tensor_load_to_lds(g0, g1, z, z, 0);
#endif
}

#define TDM_CHUNK 8   // k-tiles per chunk = 8KB per LDS buffer
__global__ void gemm_wmma_tdm(const unsigned short* Ap, const unsigned short* Wp,
                              const float* bias, const float* scale, const float* shift,
                              float* out, int ldo, int Ktiles, int Mvalid, int Nvalid,
                              int act) {
  __shared__ __align__(16) unsigned short lds_w[2][TDM_CHUNK * 512];
  int nt = blockIdx.x, mt = blockIdx.y, lane = threadIdx.x;
  int nchunks = (Ktiles + TDM_CHUNK - 1) / TDM_CHUNK;
  const unsigned short* wbase = Wp + (size_t)nt * Ktiles * 512;

  auto issue = [&](int ch) {
    int kts = Ktiles - ch * TDM_CHUNK;
    if (kts > TDM_CHUNK) kts = TDM_CHUNK;
    // flat LDS address carries the LDS byte offset in bits [31:0]
    unsigned ldsoff = (unsigned)(unsigned long long)(size_t)&lds_w[ch & 1][0];
    tdm_copy_1d(wbase + (size_t)ch * TDM_CHUNK * 512, ldsoff,
                (unsigned)kts * 1024u);
  };

  issue(0);
  floatx8 c = {};
  const unsigned short* ap = Ap + ((size_t)mt * Ktiles * 32 + lane) * 16;
  for (int ch = 0; ch < nchunks; ++ch) {
    if (ch + 1 < nchunks) {
      issue(ch + 1);                       // overlap next DMA with compute
      __builtin_amdgcn_s_wait_tensorcnt(1); // oldest (current) chunk landed
    } else {
      __builtin_amdgcn_s_wait_tensorcnt(0);
    }
    int kts = Ktiles - ch * TDM_CHUNK;
    if (kts > TDM_CHUNK) kts = TDM_CHUNK;
    const unsigned short* bp = &lds_w[ch & 1][lane * 16];
    for (int kk = 0; kk < kts; ++kk) {
      bf16x16 a = *(const bf16x16*)ap;     // A from global (L2-hot, shared)
      bf16x16 b = *(const bf16x16*)bp;     // B from LDS (ds_load_b128 x2)
      c = __builtin_amdgcn_wmma_f32_16x16x32_bf16(false, a, false, b,
                                                  (short)0, c, false, false);
      ap += 512; bp += 512;
    }
  }
  gemm_epilogue(c, nt, mt, lane, bias, scale, shift, out, ldo, Mvalid, Nvalid, act);
}
#else
#define HAVE_TDM 0
#endif

// ---------------- pre_enc = hs @ att_enc_w^T + b  (one-time, B*T*ADIM) ------
__global__ void pre_enc_kernel(const float* hs, const float* w, const float* bv,
                               float* out) {
  int idx = blockIdx.x * blockDim.x + threadIdx.x;
  if (idx >= B * T * ADIM) return;
  int a = idx % ADIM;
  int bt = idx / ADIM;
  float acc = bv[a];
  const float* hrow = hs + (size_t)bt * IDIM;
  const float* wrow = w + (size_t)a * IDIM;
  for (int d = 0; d < IDIM; ++d) acc += hrow[d] * wrow[d];
  out[idx] = acc;
}

// ---------------- location-sensitive attention energies ---------------------
__global__ void att_energy_kernel(const float* pre_enc, const float* dec,
                                  const float* convw, const float* locw,
                                  const float* vvec, const float* qv,
                                  const int* hlens, const float* att_cum,
                                  float* e, int t) {
  __shared__ float s_conv[ACONV_CH * ACONV_K];
  __shared__ float s_loc[ADIM * ACONV_CH];
  __shared__ float s_dec[ADIM];
  __shared__ float s_v[ADIM];
  int b  = blockIdx.x / (T / 128);
  int tt = (blockIdx.x % (T / 128)) * 128 + (int)threadIdx.x;
  for (int i = threadIdx.x; i < ACONV_CH * ACONV_K; i += 128) s_conv[i] = convw[i];
  for (int i = threadIdx.x; i < ADIM * ACONV_CH; i += 128)    s_loc[i]  = locw[i];
  if (threadIdx.x < ADIM) {
    s_dec[threadIdx.x] = dec[b * ADIM + threadIdx.x];
    s_v[threadIdx.x]   = vvec[threadIdx.x];
  }
  __syncthreads();
  int hlen = hlens[b];
  float inv = 1.0f / (float)hlen;
  float win[ACONV_K];
  for (int k = 0; k < ACONV_K; ++k) {
    int p = tt + k - ACONV_K / 2;
    float v = 0.f;
    if (p >= 0 && p < T)
      v = (t == 0) ? ((p < hlen) ? inv : 0.f) : att_cum[b * T + p];
    win[k] = v;
  }
  float conv[ACONV_CH];
  for (int cch = 0; cch < ACONV_CH; ++cch) {
    float acc = 0.f;
    for (int k = 0; k < ACONV_K; ++k) acc += win[k] * s_conv[cch * ACONV_K + k];
    conv[cch] = acc;
  }
  const float* pe = pre_enc + ((size_t)b * T + tt) * ADIM;
  float eacc = 0.f;
  for (int a = 0; a < ADIM; ++a) {
    float s = pe[a] + s_dec[a];
    const float* lr = s_loc + a * ACONV_CH;
    for (int cch = 0; cch < ACONV_CH; ++cch) s += conv[cch] * lr[cch];
    eacc += tanhf(s) * s_v[a];
  }
  e[b * T + tt] = (tt < hlen) ? (eacc + qv[b * T + tt]) : -1e30f;
}

// ---------------- masked softmax + att_ws store + cumulated update ----------
__global__ void att_softmax_kernel(const float* e, float* att_w, float* att_cum,
                                   float* att_ws_out, int t) {
  __shared__ float red[128];
  int b = blockIdx.x, tid = threadIdx.x;
  float m = -1e30f;
  for (int i = tid; i < T; i += 128) m = fmaxf(m, e[b * T + i]);
  red[tid] = m; __syncthreads();
  for (int s = 64; s > 0; s >>= 1) {
    if (tid < s) red[tid] = fmaxf(red[tid], red[tid + s]);
    __syncthreads();
  }
  float mx = red[0]; __syncthreads();
  float sum = 0.f;
  for (int i = tid; i < T; i += 128) sum += expf(e[b * T + i] - mx);
  red[tid] = sum; __syncthreads();
  for (int s = 64; s > 0; s >>= 1) {
    if (tid < s) red[tid] += red[tid + s];
    __syncthreads();
  }
  float inv = 1.0f / red[0];
  for (int i = tid; i < T; i += 128) {
    float w = expf(e[b * T + i] - mx) * inv;
    att_w[b * T + i] = w;
    att_ws_out[(size_t)b * L * T + (size_t)t * T + i] = w;
    att_cum[b * T + i] = (t == 0) ? w : att_cum[b * T + i] + w;
  }
}

// ---------------- context: att_c = att_w @ hs -------------------------------
__global__ void att_context_kernel(const float* att_w, const float* hs,
                                   float* att_c) {
  __shared__ float w_s[T];
  int b = blockIdx.x, tid = threadIdx.x;  // 256 threads
  for (int i = tid; i < T; i += 256) w_s[i] = att_w[b * T + i];
  __syncthreads();
  for (int d = tid; d < IDIM; d += 256) {
    float acc = 0.f;
    for (int i = 0; i < T; ++i) acc += w_s[i] * hs[((size_t)b * T + i) * IDIM + d];
    att_c[b * IDIM + d] = acc;
  }
}

// ---------------- LSTM gate nonlinearity + zoneout (eval) -------------------
__global__ void lstm_ew_kernel(const float* g, float* h, float* c) {
  int idx = blockIdx.x * blockDim.x + threadIdx.x;
  if (idx >= B * DUNITS) return;
  int b = idx / DUNITS, u = idx % DUNITS;
  const float* gb = g + (size_t)b * 4 * DUNITS;
  float gi = gb[u], gf = gb[DUNITS + u], gg = gb[2 * DUNITS + u], go = gb[3 * DUNITS + u];
  float si = 1.f / (1.f + expf(-gi));
  float sf = 1.f / (1.f + expf(-gf));
  float so = 1.f / (1.f + expf(-go));
  float cn = sf * c[idx] + si * tanhf(gg);
  float hn = so * tanhf(cn);
  h[idx] = ZONEOUT * h[idx] + (1.f - ZONEOUT) * hn;
  c[idx] = ZONEOUT * c[idx] + (1.f - ZONEOUT) * cn;
}

__global__ void bias_sum_kernel(const float* b0, const float* b1, float* out, int n) {
  int i = blockIdx.x * blockDim.x + threadIdx.x;
  if (i < n) out[i] = b0[i] + b1[i];
}

// ---------------- stop-token logit ------------------------------------------
__global__ void logit_kernel(const float* z1, const float* att_c, const float* pw,
                             const float* pb, float* out_logits, int t) {
  __shared__ float red[256];
  int b = blockIdx.x, tid = threadIdx.x;
  float acc = 0.f;
  for (int i = tid; i < DUNITS + IDIM; i += 256) {
    float x = (i < DUNITS) ? z1[b * DUNITS + i] : att_c[b * IDIM + (i - DUNITS)];
    acc += x * pw[i];
  }
  red[tid] = acc; __syncthreads();
  for (int s = 128; s > 0; s >>= 1) {
    if (tid < s) red[tid] += red[tid + s];
    __syncthreads();
  }
  if (tid == 0) out_logits[(size_t)b * L + t] = red[0] + pb[0];
}

__global__ void add_kernel(float* dst, const float* src, int n) {
  int i = blockIdx.x * blockDim.x + threadIdx.x;
  if (i < n) dst[i] += src[i];
}

// ============================================================================
extern "C" void kernel_launch(void* const* d_in, const int* in_sizes, int n_in,
                              void* d_out, int out_size, void* d_ws, size_t ws_size,
                              hipStream_t stream) {
  (void)in_sizes; (void)n_in; (void)out_size; (void)ws_size;

  const float* q_in      = (const float*)d_in[0];
  const float* hs        = (const float*)d_in[1];
  const int*   hlens     = (const int*)  d_in[2];
  const float* ys        = (const float*)d_in[3];
  const float* prenet_w1 = (const float*)d_in[4];
  const float* prenet_b1 = (const float*)d_in[5];
  const float* prenet_w2 = (const float*)d_in[6];
  const float* prenet_b2 = (const float*)d_in[7];
  const float* att_enc_w = (const float*)d_in[8];
  const float* att_enc_b = (const float*)d_in[9];
  const float* att_dec_w = (const float*)d_in[10];
  const float* att_conv_w= (const float*)d_in[11];
  const float* att_loc_w = (const float*)d_in[12];
  const float* att_v     = (const float*)d_in[13];
  const float* l0_wih    = (const float*)d_in[14];
  const float* l0_whh    = (const float*)d_in[15];
  const float* l0_bih    = (const float*)d_in[16];
  const float* l0_bhh    = (const float*)d_in[17];
  const float* l1_wih    = (const float*)d_in[18];
  const float* l1_whh    = (const float*)d_in[19];
  const float* l1_bih    = (const float*)d_in[20];
  const float* l1_bhh    = (const float*)d_in[21];
  const float* feat_w    = (const float*)d_in[22];
  const float* prob_w    = (const float*)d_in[23];
  const float* prob_b    = (const float*)d_in[24];
  const float* post_w[5], *post_g[5], *post_b[5];
  for (int i = 0; i < 5; ++i) {
    post_w[i] = (const float*)d_in[25 + i];
    post_g[i] = (const float*)d_in[30 + i];
    post_b[i] = (const float*)d_in[35 + i];
  }

  float* out_after  = (float*)d_out;
  float* out_before = out_after + (size_t)B * L * ODIM;
  float* out_logits = out_before + (size_t)B * L * ODIM;
  float* out_attws  = out_logits + (size_t)B * L;

  // --- workspace carving (256B aligned chunks) ---
  char* wsb = (char*)d_ws;
  size_t off = 0;
  auto carve = [&](size_t bytes) -> char* {
    char* p = wsb + off;
    off += (bytes + 255) & ~(size_t)255;
    return p;
  };
  unsigned short* wp_l0 = (unsigned short*)carve((size_t)256 * 56 * 512 * 2); // (4096,1792)
  unsigned short* wp_l1 = (unsigned short*)carve((size_t)256 * 64 * 512 * 2); // (4096,2048)
  unsigned short* wp_dec= (unsigned short*)carve((size_t)8   * 32 * 512 * 2); // (128,1024)
  unsigned short* wp_p1 = (unsigned short*)carve((size_t)16  * 3  * 512 * 2); // (256,80->96)
  unsigned short* wp_p2 = (unsigned short*)carve((size_t)16  * 8  * 512 * 2); // (256,256)
  unsigned short* wp_ft = (unsigned short*)carve((size_t)5   * 48 * 512 * 2); // (80,1536)
  unsigned short* wp_c1 = (unsigned short*)carve((size_t)32  * 13 * 512 * 2); // (512,400)
  unsigned short* wp_c2 = (unsigned short*)carve((size_t)32  * 80 * 512 * 2); // (512,2560)
  unsigned short* wp_c3 = (unsigned short*)carve((size_t)32  * 80 * 512 * 2);
  unsigned short* wp_c4 = (unsigned short*)carve((size_t)32  * 80 * 512 * 2);
  unsigned short* wp_c5 = (unsigned short*)carve((size_t)5   * 80 * 512 * 2); // (80,2560)
  float* bsum0   = (float*)carve(4096 * 4);
  float* bsum1   = (float*)carve(4096 * 4);
  float* pre_enc = (float*)carve((size_t)B * T * ADIM * 4);
  float* state   = (float*)carve((size_t)4 * B * DUNITS * 4);  // z0|c0|z1|c1
  float* z0 = state, *c0 = z0 + B * DUNITS, *z1 = c0 + B * DUNITS, *c1 = z1 + B * DUNITS;
  float* att_cum = (float*)carve(B * T * 4);
  float* e_buf   = (float*)carve(B * T * 4);
  float* att_w   = (float*)carve(B * T * 4);
  float* att_c   = (float*)carve(B * IDIM * 4);
  float* dec_buf = (float*)carve(B * ADIM * 4);
  float* pre1    = (float*)carve(B * PRENET_D * 4);
  float* pre2    = (float*)carve(B * PRENET_D * 4);
  float* gates   = (float*)carve((size_t)B * 4 * DUNITS * 4);
  unsigned short* a_dec = (unsigned short*)carve((size_t)32 * 512 * 2);
  unsigned short* a_p1  = (unsigned short*)carve((size_t)3  * 512 * 2);
  unsigned short* a_p2  = (unsigned short*)carve((size_t)8  * 512 * 2);
  unsigned short* a_l0  = (unsigned short*)carve((size_t)56 * 512 * 2);
  unsigned short* a_l1  = (unsigned short*)carve((size_t)64 * 512 * 2);
  unsigned short* a_ft  = (unsigned short*)carve((size_t)48 * 512 * 2);
  unsigned short* a_post= (unsigned short*)carve((size_t)96 * 80 * 512 * 2);
  float* pbuf_a = (float*)carve((size_t)B * L * POST_CH * 4);
  float* pbuf_b = (float*)carve((size_t)B * L * POST_CH * 4);

  // --- per-launch setup: zero state, pack weights, fused biases, pre_enc ---
  hipMemsetAsync(state, 0, (size_t)4 * B * DUNITS * 4, stream);
  pack_w<<<dim3(56, 256), 32, 0, stream>>>(l0_wih, 768,  l0_whh, 1024, 4096, 56, wp_l0);
  pack_w<<<dim3(64, 256), 32, 0, stream>>>(l1_wih, 1024, l1_whh, 1024, 4096, 64, wp_l1);
  pack_w<<<dim3(32, 8),   32, 0, stream>>>(att_dec_w, 1024, nullptr, 0, 128, 32, wp_dec);
  pack_w<<<dim3(3, 16),   32, 0, stream>>>(prenet_w1, 80,   nullptr, 0, 256, 3,  wp_p1);
  pack_w<<<dim3(8, 16),   32, 0, stream>>>(prenet_w2, 256,  nullptr, 0, 256, 8,  wp_p2);
  pack_w<<<dim3(48, 5),   32, 0, stream>>>(feat_w, 1536,    nullptr, 0, 80,  48, wp_ft);
  pack_w<<<dim3(13, 32),  32, 0, stream>>>(post_w[0], ODIM * POST_K,    nullptr, 0, POST_CH, 13, wp_c1);
  pack_w<<<dim3(80, 32),  32, 0, stream>>>(post_w[1], POST_CH * POST_K, nullptr, 0, POST_CH, 80, wp_c2);
  pack_w<<<dim3(80, 32),  32, 0, stream>>>(post_w[2], POST_CH * POST_K, nullptr, 0, POST_CH, 80, wp_c3);
  pack_w<<<dim3(80, 32),  32, 0, stream>>>(post_w[3], POST_CH * POST_K, nullptr, 0, POST_CH, 80, wp_c4);
  pack_w<<<dim3(80, 5),   32, 0, stream>>>(post_w[4], POST_CH * POST_K, nullptr, 0, ODIM,    80, wp_c5);
  bias_sum_kernel<<<16, 256, 0, stream>>>(l0_bih, l0_bhh, bsum0, 4096);
  bias_sum_kernel<<<16, 256, 0, stream>>>(l1_bih, l1_bhh, bsum1, 4096);
  pre_enc_kernel<<<(B * T * ADIM + 255) / 256, 256, 0, stream>>>(hs, att_enc_w, att_enc_b, pre_enc);

#if HAVE_TDM
#define BIG_GEMM gemm_wmma_tdm
#else
#define BIG_GEMM gemm_wmma
#endif

  // --- sequential decoder scan ---
  for (int t = 0; t < L; ++t) {
    // dec = z0_old @ att_dec_w^T
    pack_act8<<<32, 32, 0, stream>>>(z0, DUNITS, DUNITS, nullptr, 0, 0, nullptr, 0, 0, a_dec, 0);
    gemm_wmma<<<dim3(8, 1), 32, 0, stream>>>(a_dec, wp_dec, nullptr, nullptr, nullptr,
                                             dec_buf, ADIM, 32, B, ADIM, 0);
    // energies, softmax (+att_ws store, +cumulation), context
    att_energy_kernel<<<B * (T / 128), 128, 0, stream>>>(pre_enc, dec_buf, att_conv_w,
                                                         att_loc_w, att_v, q_in, hlens,
                                                         att_cum, e_buf, t);
    att_softmax_kernel<<<B, 128, 0, stream>>>(e_buf, att_w, att_cum, out_attws, t);
    att_context_kernel<<<B, 256, 0, stream>>>(att_w, hs, att_c);
    // prenet (teacher forcing: y_{t-1}, zeros at t=0)
    pack_act8<<<3, 32, 0, stream>>>(ys + (size_t)(t > 0 ? t - 1 : 0) * ODIM, ODIM, L * ODIM,
                                    nullptr, 0, 0, nullptr, 0, 0, a_p1, t == 0 ? 1 : 0);
    gemm_wmma<<<dim3(16, 1), 32, 0, stream>>>(a_p1, wp_p1, prenet_b1, nullptr, nullptr,
                                              pre1, PRENET_D, 3, B, PRENET_D, 1);
    pack_act8<<<8, 32, 0, stream>>>(pre1, PRENET_D, PRENET_D, nullptr, 0, 0, nullptr, 0, 0, a_p2, 0);
    gemm_wmma<<<dim3(16, 1), 32, 0, stream>>>(a_p2, wp_p2, prenet_b2, nullptr, nullptr,
                                              pre2, PRENET_D, 8, B, PRENET_D, 1);
    // LSTM0: gates = [att_c|pre2|z0] @ [Wih|Whh]^T + (bih+bhh)   (TDM-streamed)
    pack_act8<<<56, 32, 0, stream>>>(att_c, IDIM, IDIM, pre2, PRENET_D, PRENET_D,
                                     z0, DUNITS, DUNITS, a_l0, 0);
    BIG_GEMM<<<dim3(256, 1), 32, 0, stream>>>(a_l0, wp_l0, bsum0, nullptr, nullptr,
                                              gates, 4 * DUNITS, 56, B, 4 * DUNITS, 0);
    lstm_ew_kernel<<<(B * DUNITS) / 256, 256, 0, stream>>>(gates, z0, c0);
    // LSTM1: gates = [z0_new|z1_old] @ [Wih|Whh]^T + (bih+bhh)   (TDM-streamed)
    pack_act8<<<64, 32, 0, stream>>>(z0, DUNITS, DUNITS, z1, DUNITS, DUNITS,
                                     nullptr, 0, 0, a_l1, 0);
    BIG_GEMM<<<dim3(256, 1), 32, 0, stream>>>(a_l1, wp_l1, bsum1, nullptr, nullptr,
                                              gates, 4 * DUNITS, 64, B, 4 * DUNITS, 0);
    lstm_ew_kernel<<<(B * DUNITS) / 256, 256, 0, stream>>>(gates, z1, c1);
    // outputs: before[:,t,:] = [z1|att_c] @ feat_w^T ; stop logit
    pack_act8<<<48, 32, 0, stream>>>(z1, DUNITS, DUNITS, att_c, IDIM, IDIM,
                                     nullptr, 0, 0, a_ft, 0);
    gemm_wmma<<<dim3(5, 1), 32, 0, stream>>>(a_ft, wp_ft, nullptr, nullptr, nullptr,
                                             out_before + (size_t)t * ODIM, L * ODIM,
                                             48, B, ODIM, 0);
    logit_kernel<<<B, 256, 0, stream>>>(z1, att_c, prob_w, prob_b, out_logits, t);
  }

  // --- postnet: 5 conv layers as implicit GEMM (M=B*L=1536 rows, TDM weights)
  pack_conv_act<<<dim3(13, 96), 32, 0, stream>>>(out_before, ODIM, 13, a_post);
  BIG_GEMM<<<dim3(32, 96), 32, 0, stream>>>(a_post, wp_c1, nullptr, post_g[0], post_b[0],
                                            pbuf_a, POST_CH, 13, B * L, POST_CH, 2);
  pack_conv_act<<<dim3(80, 96), 32, 0, stream>>>(pbuf_a, POST_CH, 80, a_post);
  BIG_GEMM<<<dim3(32, 96), 32, 0, stream>>>(a_post, wp_c2, nullptr, post_g[1], post_b[1],
                                            pbuf_b, POST_CH, 80, B * L, POST_CH, 2);
  pack_conv_act<<<dim3(80, 96), 32, 0, stream>>>(pbuf_b, POST_CH, 80, a_post);
  BIG_GEMM<<<dim3(32, 96), 32, 0, stream>>>(a_post, wp_c3, nullptr, post_g[2], post_b[2],
                                            pbuf_a, POST_CH, 80, B * L, POST_CH, 2);
  pack_conv_act<<<dim3(80, 96), 32, 0, stream>>>(pbuf_a, POST_CH, 80, a_post);
  BIG_GEMM<<<dim3(32, 96), 32, 0, stream>>>(a_post, wp_c4, nullptr, post_g[3], post_b[3],
                                            pbuf_b, POST_CH, 80, B * L, POST_CH, 2);
  pack_conv_act<<<dim3(80, 96), 32, 0, stream>>>(pbuf_b, POST_CH, 80, a_post);
  BIG_GEMM<<<dim3(5, 96), 32, 0, stream>>>(a_post, wp_c5, nullptr, post_g[4], post_b[4],
                                           out_after, ODIM, 80, B * L, ODIM, 0);
  // after = before + postnet(before)
  add_kernel<<<(B * L * ODIM + 255) / 256, 256, 0, stream>>>(out_after, out_before,
                                                             B * L * ODIM);
}
